// SpModel_45277545234711
// MI455X (gfx1250) — matrix operand order repaired and checked by reading, not compiled, not run on hardware
//
#include <hip/hip_runtime.h>
#include <hip/hip_bf16.h>

#define H 128
#define BSTRIDE   136   // halfs per LDS row for W^T (pad breaks 256B bank alignment)
#define ASTRIDE_F 136   // floats per LDS row for A tile (544B rows, 16B aligned)

typedef __attribute__((ext_vector_type(16))) __bf16 v16bf;
typedef __attribute__((ext_vector_type(8)))  __bf16 v8bf;
typedef __attribute__((ext_vector_type(4)))  __bf16 v4bf;
typedef __attribute__((ext_vector_type(8)))  float  v8f;

// Assemble a 16-half WMMA B fragment from two 16B LDS loads `gap` halfs apart.
__device__ __forceinline__ v16bf ld_frag_bf(const __bf16* p, int gap) {
  v8bf lo = *reinterpret_cast<const v8bf*>(p);
  v8bf hi = *reinterpret_cast<const v8bf*>(p + gap);
  v16bf r;
#pragma unroll
  for (int i = 0; i < 8; ++i) { r[i] = lo[i]; r[i + 8] = hi[i]; }
  return r;
}

// Assemble a 16-half WMMA A fragment from 16 f32 values in LDS (8 at p, 8 at p+16),
// converting f32 -> bf16 in registers.
__device__ __forceinline__ v16bf ld_frag_f32(const float* p) {
  float4 f0 = *reinterpret_cast<const float4*>(p);
  float4 f1 = *reinterpret_cast<const float4*>(p + 4);
  float4 f2 = *reinterpret_cast<const float4*>(p + 16);
  float4 f3 = *reinterpret_cast<const float4*>(p + 20);
  v16bf r;
  r[0]  = (__bf16)f0.x; r[1]  = (__bf16)f0.y; r[2]  = (__bf16)f0.z; r[3]  = (__bf16)f0.w;
  r[4]  = (__bf16)f1.x; r[5]  = (__bf16)f1.y; r[6]  = (__bf16)f1.z; r[7]  = (__bf16)f1.w;
  r[8]  = (__bf16)f2.x; r[9]  = (__bf16)f2.y; r[10] = (__bf16)f2.z; r[11] = (__bf16)f2.w;
  r[12] = (__bf16)f3.x; r[13] = (__bf16)f3.y; r[14] = (__bf16)f3.z; r[15] = (__bf16)f3.w;
  return r;
}

// float atomic max via int-max (val>=0) / uint-min (val<0). Identity: -NaN (0xFFFFFFFF).
__device__ __forceinline__ void atomicMaxF(float* addr, float val) {
  if (val >= 0.0f) atomicMax(reinterpret_cast<int*>(addr), __float_as_int(val));
  else             atomicMin(reinterpret_cast<unsigned int*>(addr), __float_as_uint(val));
}

// ---------------- K0: tbl0 = x_emb@W0 + b0, tbl1 = x_emb@W1 + b1 (32x128 each) ----
__global__ void k_tables(const float* __restrict__ x_emb,
                         const float* __restrict__ W0, const float* __restrict__ b0,
                         const float* __restrict__ W1, const float* __restrict__ b1,
                         float* __restrict__ tbl0, float* __restrict__ tbl1) {
  int c = blockIdx.x;    // 0..31
  int h = threadIdx.x;   // 0..127
  float a0 = b0[h], a1 = b1[h];
#pragma unroll 4
  for (int k = 0; k < H; ++k) {
    float xv = x_emb[c * H + k];
    a0 += xv * W0[k * H + h];
    a1 += xv * W1[k * H + h];
  }
  tbl0[c * H + h] = a0;
  tbl1[c * H + h] = a1;
}

// ---------------- K1: Xv[m] = tbl0[ci]*tbl1[cj]*tup_emb[ct]; + bf16 shadow -------
__global__ void k_tupleinit(const int* __restrict__ tuple_codes,
                            const int* __restrict__ tuple_i,
                            const int* __restrict__ tuple_j,
                            const int* __restrict__ x_codes,
                            const float* __restrict__ tup_emb,
                            const float* __restrict__ tbl0,
                            const float* __restrict__ tbl1,
                            float* __restrict__ Xv32, __bf16* __restrict__ Xvb, int M) {
  int m = blockIdx.x * 8 + (threadIdx.x >> 5);
  if (m >= M) return;
  int lane = threadIdx.x & 31;
  int h = lane * 4;
  int ci = x_codes[tuple_i[m]];
  int cj = x_codes[tuple_j[m]];
  int ct = tuple_codes[m];
  float4 a = *reinterpret_cast<const float4*>(tbl0 + ci * H + h);
  float4 b = *reinterpret_cast<const float4*>(tbl1 + cj * H + h);
  float4 t = *reinterpret_cast<const float4*>(tup_emb + ct * H + h);
  float4 v;
  v.x = a.x * b.x * t.x; v.y = a.y * b.y * t.y;
  v.z = a.z * b.z * t.z; v.w = a.w * b.w * t.w;
  size_t o = (size_t)m * H + h;
  *reinterpret_cast<float4*>(Xv32 + o) = v;
  v4bf vb; vb[0] = (__bf16)v.x; vb[1] = (__bf16)v.y; vb[2] = (__bf16)v.z; vb[3] = (__bf16)v.w;
  *reinterpret_cast<v4bf*>(Xvb + o) = vb;
}

// ---------------- K2: msg = Xvb[src]*ea; scatter-add into agg (f32 atomics) ------
__global__ void k_msg(const __bf16* __restrict__ Xvb, const float* __restrict__ ea_emb,
                      const int* __restrict__ msg_src, const int* __restrict__ msg_dst,
                      const int* __restrict__ msg_ea, float* __restrict__ agg, int ME) {
  int e = blockIdx.x * 8 + (threadIdx.x >> 5);
  if (e >= ME) return;
  int lane = threadIdx.x & 31;
  int h = lane * 4;
  int s = msg_src[e], d = msg_dst[e], a = msg_ea[e];
  v4bf xb = *reinterpret_cast<const v4bf*>(Xvb + (size_t)s * H + h);
  float4 ea = *reinterpret_cast<const float4*>(ea_emb + a * H + h);
  float* ag = agg + (size_t)d * H + h;
  atomicAdd(ag + 0, (float)xb[0] * ea.x);
  atomicAdd(ag + 1, (float)xb[1] * ea.y);
  atomicAdd(ag + 2, (float)xb[2] * ea.z);
  atomicAdd(ag + 3, (float)xb[3] * ea.w);
}

// ---------------- K3: Xv += relu(agg @ Wc[l] + bc[l])  (bf16 WMMA, f32 acc) ------
// Block: 256 threads (8 waves), 64 rows x 128 cols of output.
// W staged transposed as bf16 (B frags contiguous per lane); A tile staged as raw
// f32 via GLOBAL_LOAD_ASYNC_TO_LDS_B128 (ASYNCcnt), converted during frag build.
__global__ void __launch_bounds__(256)
k_conv_gemm(const float* __restrict__ agg, const float* __restrict__ Wc_l,
            const float* __restrict__ bc_l,
            float* __restrict__ Xv32, __bf16* __restrict__ Xvb, int M) {
  __shared__ __align__(16) __bf16 sW[H * BSTRIDE];      // sW[n*BSTRIDE + k] = Wc[k][n]
  __shared__ __align__(16) float  sA[64 * ASTRIDE_F];   // sA[r*ASTRIDE_F + k] (f32)
  int tid = threadIdx.x;
  int rowbase = blockIdx.x * 64;
  bool full = (rowbase + 64 <= M);

  for (int idx = tid; idx < H * H; idx += 256) {
    int k = idx >> 7, n = idx & (H - 1);
    sW[n * BSTRIDE + k] = (__bf16)Wc_l[idx];            // transpose + cvt on the fly
  }

  if (full) {
    // 64 rows x 512B = 32KB, copied as 2048 x 16B chunks; 8 async instrs per wave.
    uint32_t abase = (uint32_t)(uintptr_t)(void*)&sA[0];  // generic low 32b = LDS offset
    uint64_t gbase = (uint64_t)(uintptr_t)(agg + (size_t)rowbase * H);
#pragma unroll
    for (int it = 0; it < 8; ++it) {
      int idx = tid + it * 256;                  // chunk id 0..2047
      int r = idx >> 5, c = idx & 31;            // row, 16B-chunk within row
      uint32_t lofs = abase + (uint32_t)(r * (ASTRIDE_F * 4) + c * 16);
      uint32_t gofs = (uint32_t)(idx << 4);      // global rows are dense (512B)
      asm volatile("global_load_async_to_lds_b128 %0, %1, %2 offset:0"
                   :: "v"(lofs), "v"(gofs), "s"(gbase) : "memory");
    }
    asm volatile("s_wait_asynccnt 0x0" ::: "memory");
  } else {
    for (int idx = tid; idx < 64 * H; idx += 256) {
      int r = idx >> 7, k = idx & (H - 1);
      int row = rowbase + r;
      sA[r * ASTRIDE_F + k] = (row < M) ? agg[(size_t)row * H + k] : 0.0f;
    }
  }
  __syncthreads();

  int wave = tid >> 5;
  int lane = tid & 31;
  int hi   = lane >> 4;    // half-wave: 0 or 1
  int nl   = lane & 15;
  int rt   = wave & 3;     // row tile 0..3  (rows rt*16..rt*16+15)
  int cg   = wave >> 2;    // column group 0/1; this wave does col tiles 2c+cg

  // A fragments (ISA layout: lanes0-15 -> K kb+0..7,kb+16..23; lanes16-31 -> +8)
  v16bf afrag[4];
#pragma unroll
  for (int kk = 0; kk < 4; ++kk) {
    const float* ap = &sA[(rt * 16 + nl) * ASTRIDE_F + kk * 32 + hi * 8];
    afrag[kk] = ld_frag_f32(ap);
  }

#pragma unroll
  for (int c = 0; c < 4; ++c) {
    int ct  = c * 2 + cg;
    int col = ct * 16 + nl;
    v8f acc = {};
#pragma unroll
    for (int kk = 0; kk < 4; ++kk) {
      // B layout: lanes0-15 -> K kb+0..7,kb+16..23; lanes16-31 -> +8 (matches A)
      const __bf16* bp = &sW[(ct * 16 + nl) * BSTRIDE + kk * 32 + hi * 8];
      v16bf bfrag = ld_frag_bf(bp, 16);
      acc = __builtin_amdgcn_wmma_f32_16x16x32_bf16(
          false, afrag[kk], false, bfrag, (short)0, acc, false, false);
    }
    float bias = bc_l[col];
    if (full) {
#pragma unroll
      for (int r = 0; r < 8; ++r) {
        int row = rowbase + rt * 16 + hi * 8 + r;  // C layout: VGPR r -> M=r / M=r+8
        float v = acc[r] + bias;
        v = v > 0.0f ? v : 0.0f;
        size_t gi = (size_t)row * H + col;
        float nv = Xv32[gi] + v;
        Xv32[gi] = nv;
        Xvb[gi]  = (__bf16)nv;
      }
    } else {
#pragma unroll
      for (int r = 0; r < 8; ++r) {
        int row = rowbase + rt * 16 + hi * 8 + r;
        if (row < M) {
          float v = acc[r] + bias;
          v = v > 0.0f ? v : 0.0f;
          size_t gi = (size_t)row * H + col;
          float nv = Xv32[gi] + v;
          Xv32[gi] = nv;
          Xvb[gi]  = (__bf16)nv;
        }
      }
    }
  }
}

// ---------------- K4: xp[n] = segment_max(Xv, tuple_i) --------------------------
__global__ void k_poolmax(const float* __restrict__ Xv32, const int* __restrict__ tuple_i,
                          float* __restrict__ xp, int M) {
  int m = blockIdx.x * 8 + (threadIdx.x >> 5);
  if (m >= M) return;
  int lane = threadIdx.x & 31;
  int h = lane * 4;
  int n = tuple_i[m];
  float4 v = *reinterpret_cast<const float4*>(Xv32 + (size_t)m * H + h);
  float* o = xp + (size_t)n * H + h;
  atomicMaxF(o + 0, v.x); atomicMaxF(o + 1, v.y);
  atomicMaxF(o + 2, v.z); atomicMaxF(o + 3, v.w);
}

// ---------------- K5: hs[g] += finite(xp[n]); cnt[g] += 1 -----------------------
__global__ void k_graphsum(const float* __restrict__ xp, const int* __restrict__ batch,
                           float* __restrict__ hs, float* __restrict__ cnt, int N) {
  int n = blockIdx.x * 8 + (threadIdx.x >> 5);
  if (n >= N) return;
  int lane = threadIdx.x & 31;
  int h = lane * 4;
  int b = batch[n];
  float4 v = *reinterpret_cast<const float4*>(xp + (size_t)n * H + h);
  float x0 = isfinite(v.x) ? v.x : 0.0f;
  float x1 = isfinite(v.y) ? v.y : 0.0f;
  float x2 = isfinite(v.z) ? v.z : 0.0f;
  float x3 = isfinite(v.w) ? v.w : 0.0f;
  float* o = hs + (size_t)b * H + h;
  atomicAdd(o + 0, x0); atomicAdd(o + 1, x1);
  atomicAdd(o + 2, x2); atomicAdd(o + 3, x3);
  if (lane == 0) atomicAdd(&cnt[b], 1.0f);
}

// ---------------- K6: out[g] = (hs[g]/max(cnt,1)) . Wp + bp ---------------------
__global__ void k_out(const float* __restrict__ hs, const float* __restrict__ cnt,
                      const float* __restrict__ Wp, const float* __restrict__ bp,
                      float* __restrict__ out) {
  __shared__ float red[H];
  int g = blockIdx.x, h = threadIdx.x;
  float c = cnt[g]; c = c > 1.0f ? c : 1.0f;
  red[h] = (hs[(size_t)g * H + h] / c) * Wp[h];
  __syncthreads();
  for (int s = H / 2; s > 0; s >>= 1) {
    if (h < s) red[h] += red[h + s];
    __syncthreads();
  }
  if (h == 0) out[g] = red[0] + bp[0];
}

extern "C" void kernel_launch(void* const* d_in, const int* in_sizes, int n_in,
                              void* d_out, int out_size, void* d_ws, size_t ws_size,
                              hipStream_t stream) {
  const int*   x_codes     = (const int*)d_in[0];
  const int*   tuple_codes = (const int*)d_in[1];
  const int*   tuple_i     = (const int*)d_in[2];
  const int*   tuple_j     = (const int*)d_in[3];
  const int*   msg_src     = (const int*)d_in[4];
  const int*   msg_dst     = (const int*)d_in[5];
  const int*   msg_ea      = (const int*)d_in[6];
  const int*   batch       = (const int*)d_in[7];
  const float* x_emb       = (const float*)d_in[9];
  const float* tup_emb     = (const float*)d_in[10];
  const float* ea_emb      = (const float*)d_in[11];
  const float* W0          = (const float*)d_in[12];
  const float* b0          = (const float*)d_in[13];
  const float* W1          = (const float*)d_in[14];
  const float* b1          = (const float*)d_in[15];
  const float* Wc          = (const float*)d_in[16];
  const float* bc          = (const float*)d_in[17];
  const float* Wp          = (const float*)d_in[18];
  const float* bp          = (const float*)d_in[19];
  float* out = (float*)d_out;

  const int N  = in_sizes[0];
  const int M  = in_sizes[1];
  const int ME = in_sizes[4];
  const int G  = out_size;               // T == 1
  const int L  = in_sizes[16] / (H * H); // 5

  char* w = (char*)d_ws;
  float*  tbl0 = (float*)w;  w += (size_t)32 * H * 4;
  float*  tbl1 = (float*)w;  w += (size_t)32 * H * 4;
  float*  Xv32 = (float*)w;  w += (size_t)M * H * 4;
  __bf16* Xvb  = (__bf16*)w; w += (size_t)M * H * 2;
  float*  agg  = (float*)w;  w += (size_t)M * H * 4;
  float*  xp   = (float*)w;  w += (size_t)N * H * 4;
  float*  hs   = (float*)w;  w += (size_t)G * H * 4;
  float*  cnt  = (float*)w;  w += (size_t)G * 4;

  // input encoding folded into 32x128 tables (x has only 32 distinct rows)
  k_tables<<<32, 128, 0, stream>>>(x_emb, W0, b0, W1, b1, tbl0, tbl1);
  k_tupleinit<<<(M + 7) / 8, 256, 0, stream>>>(tuple_codes, tuple_i, tuple_j, x_codes,
                                               tup_emb, tbl0, tbl1, Xv32, Xvb, M);
  for (int l = 0; l < L; ++l) {
    hipMemsetAsync(agg, 0, (size_t)M * H * 4, stream);
    k_msg<<<(ME + 7) / 8, 256, 0, stream>>>(Xvb, ea_emb, msg_src, msg_dst, msg_ea, agg, ME);
    k_conv_gemm<<<(M + 63) / 64, 256, 0, stream>>>(agg, Wc + (size_t)l * H * H,
                                                   bc + (size_t)l * H, Xv32, Xvb, M);
  }
  hipMemsetAsync(xp, 0xFF, (size_t)N * H * 4, stream);   // -NaN: identity for atomicMaxF
  hipMemsetAsync(hs, 0, (size_t)G * H * 4, stream);
  hipMemsetAsync(cnt, 0, (size_t)G * 4, stream);
  k_poolmax<<<(M + 7) / 8, 256, 0, stream>>>(Xv32, tuple_i, xp, M);
  k_graphsum<<<(N + 7) / 8, 256, 0, stream>>>(xp, batch, hs, cnt, N);
  k_out<<<G, H, 0, stream>>>(hs, cnt, Wp, bp, out);
}
